// ResidualBlockDynamic_1x9_71279277244556
// MI455X (gfx1250) — matrix operand order repaired and checked by reading, not compiled, not run on hardware
//
#include <hip/hip_runtime.h>
#include <hip/hip_bf16.h>
#include <math.h>

// ---------------------------------------------------------------------------
// ResidualBlockDynamic (DDFPack + LeakyReLU + 1x9 conv + residual) for gfx1250
// B=4, C=64, H=W=192, TAPS=9, mid=12.
//
// Heavy op (1x9 conv, K=576) runs on bf16 WMMA 16x16x32 with f32 accumulate.
// Intermediate 'a' is written BHWC bf16 (async LDS->global stores) so
// B-fragments are channel-contiguous in LDS (two aligned ds_load_b128 per
// lane per WMMA). The GEMM kernel stages its row with async global->LDS
// loads. Each wave owns one M-tile and six N-tiles: six live accumulators,
// each A fragment loaded once per K-step and fed to six independent WMMAs
// (no D->A/B hazards between them, minimal weight traffic, no spills).
// ---------------------------------------------------------------------------

typedef __attribute__((ext_vector_type(16))) __bf16 v16bf;
typedef __attribute__((ext_vector_type(8)))  float  v8f;

#define BB 4
#define CC 64
#define HH 192
#define WW 192
#define HW (HH * WW)          // 36864
#define MID 12
#define TAPS 9
#define STDC 0.47140452079103173f   // sqrt(2)/3

// workspace layout:
//   g     : 256 f32                       @ 0
//   cf    : 2304 f32                      @ 1024 B
//   wfrag : 36864 bf16 (73728 B)          @ 10240 B
//   a_ws  : 4*192*192*64 bf16 (18.87 MB)  @ 83968 B

// ------------------------- kernel 1: global average pool -------------------
__global__ void k_gap(const float* __restrict__ x, float* __restrict__ g) {
  int bc = blockIdx.x;               // 0..255  (b*64 + c)
  int tid = threadIdx.x;             // 256 threads
  const float* p = x + (size_t)bc * HW;
  float s = 0.f;
  for (int i = tid; i < HW; i += 256) s += p[i];
  __shared__ float red[256];
  red[tid] = s;
  __syncthreads();
  for (int off = 128; off > 0; off >>= 1) {
    if (tid < off) red[tid] += red[tid + off];
    __syncthreads();
  }
  if (tid == 0) g[bc] = red[0] * (1.0f / (float)HW);
}

// ------------------------- kernel 2: channel-filter MLP + FilterNorm -------
__global__ void k_cf(const float* __restrict__ g,
                     const float* __restrict__ cw1, const float* __restrict__ cb1,
                     const float* __restrict__ cw2, const float* __restrict__ cb2,
                     float* __restrict__ cf) {
  __shared__ float gs[BB * CC];
  __shared__ float h1[BB][MID];
  int tid = threadIdx.x;             // 256 threads, single block
  gs[tid] = g[tid];
  __syncthreads();
  if (tid < BB * MID) {
    int b = tid / MID, m = tid % MID;
    float acc = cb1[m];
    #pragma unroll 8
    for (int c = 0; c < CC; ++c) acc += gs[b * CC + c] * cw1[m * CC + c];
    h1[b][m] = acc > 0.f ? acc : 0.f;
  }
  __syncthreads();
  int b = tid / CC, c = tid % CC;
  float f[TAPS];
  float sum = 0.f, sumsq = 0.f;
  #pragma unroll
  for (int t = 0; t < TAPS; ++t) {
    int r = c * TAPS + t;
    float acc = cb2[r];
    #pragma unroll
    for (int m = 0; m < MID; ++m) acc += h1[b][m] * cw2[r * MID + m];
    f[t] = acc; sum += acc; sumsq += acc * acc;
  }
  float mean = sum * (1.0f / 9.0f);
  float var  = (sumsq - sum * sum * (1.0f / 9.0f)) * (1.0f / 8.0f);  // ddof=1
  var = var < 0.f ? 0.f : var;
  float inv = STDC / (sqrtf(var) + 1e-10f);
  #pragma unroll
  for (int t = 0; t < TAPS; ++t)
    cf[(b * CC + c) * TAPS + t] = (f[t] - mean) * inv;
}

// ------------------------- kernel 3: pack w2 into WMMA A-fragments ---------
// wfrag[((k*4+mt)*2+ks2)*32 + lane][j] = bf16(w2[m][c][k])
//   m = mt*16 + lane%16; half = lane/16
//   c = ks2*32 + half*16 + j                  (contiguous k order, matches B)
__global__ void k_wpack(const float* __restrict__ w2, __bf16* __restrict__ wfrag) {
  int t = blockIdx.x * 256 + threadIdx.x;    // 36864 = 9<<12
  int j    = t & 15;
  int lane = (t >> 4) & 31;
  int ks2  = (t >> 9) & 1;
  int mt   = (t >> 10) & 3;
  int k    = t >> 12;                        // 0..8
  int half = lane >> 4;
  int m    = mt * 16 + (lane & 15);
  int jj   = j;
  int c    = ks2 * 32 + half * 16 + jj;
  wfrag[t] = (__bf16)w2[(m * CC + c) * TAPS + k];
}

// ------------------------- kernel 4: spatial filter + dynamic DW 3x3 -------
// One block per (b,h); 192 threads (thread = w). Emits a[b,h,w,c] bf16 (BHWC)
// through LDS transpose + async LDS->global b128 stores.
__global__ void k_ddf(const float* __restrict__ x,
                      const float* __restrict__ sw, const float* __restrict__ sb,
                      const float* __restrict__ cf,
                      __bf16* __restrict__ a_ws) {
  int bh = blockIdx.x;
  int b = bh / HH, h = bh % HH;
  int w = threadIdx.x;                       // 0..191
  __shared__ __align__(16) __bf16 trans[WW][72];   // stride 72 -> 144B rows
  const float* xb = x + (size_t)b * CC * HW;

  // phase 1: spatial filters s[t](w) = sum_c x*sw + sb, then FilterNorm
  float s[TAPS];
  #pragma unroll
  for (int t = 0; t < TAPS; ++t) s[t] = sb[t];
  for (int c = 0; c < CC; ++c) {
    float xv = xb[(c * HH + h) * WW + w];
    #pragma unroll
    for (int t = 0; t < TAPS; ++t) s[t] += xv * sw[t * CC + c];
  }
  float sum = 0.f, sumsq = 0.f;
  #pragma unroll
  for (int t = 0; t < TAPS; ++t) { sum += s[t]; sumsq += s[t] * s[t]; }
  float mean = sum * (1.0f / 9.0f);
  float var  = (sumsq - sum * sum * (1.0f / 9.0f)) * (1.0f / 8.0f);
  var = var < 0.f ? 0.f : var;
  float inv = STDC / (sqrtf(var) + 1e-10f);
  #pragma unroll
  for (int t = 0; t < TAPS; ++t) s[t] = (s[t] - mean) * inv;

  // phase 2: dynamic depthwise 3x3 (mul-combined filters) + LeakyReLU(0.1)
  const float* cfb = cf + (size_t)b * CC * TAPS;
  for (int c = 0; c < CC; ++c) {
    const float* xc = xb + c * HW + h * WW + w;
    const float* cfc = cfb + c * TAPS;
    float o = 0.f;
    #pragma unroll
    for (int t = 0; t < TAPS; ++t) {
      int di = t / 3 - 1, dj = t % 3 - 1;
      int hh = h + di, ww2 = w + dj;
      float xv = (hh >= 0 && hh < HH && ww2 >= 0 && ww2 < WW)
                     ? xc[di * WW + dj] : 0.f;
      o += xv * s[t] * cfc[t];
    }
    o = o >= 0.f ? o : 0.1f * o;
    trans[w][c] = (__bf16)o;
  }
  __syncthreads();

  // coalesced BHWC store via async LDS->global: 1536 x 16B chunks
  uint4* dstg = (uint4*)(a_ws + (size_t)bh * (WW * CC));
  const char* base = (const char*)&trans[0][0];
  for (int q = w; q < 1536; q += WW) {
    int wr = q >> 3, part = q & 7;
    uint32_t loff = (uint32_t)(uintptr_t)(base + wr * 144 + part * 16);
    uint4* gp = dstg + q;
    asm volatile("global_store_async_from_lds_b128 %0, %1, off"
                 :: "v"(gp), "v"(loff) : "memory");
  }
  asm volatile("s_wait_asynccnt 0x0" ::: "memory");
}

// ------------------------- kernel 5: 1x9 conv as 9 bf16 WMMA GEMMs ---------
// One block per (b,h); 256 threads = 8 waves. Wave owns mt = wave/2 and six
// nt tiles; six accumulators stay live, each A fragment is loaded once per
// K-step and feeds six independent WMMAs.
__global__ void k_conv9(const __bf16* __restrict__ a_ws,
                        const __bf16* __restrict__ wfrag,
                        const float* __restrict__ x,
                        const float* __restrict__ b2,
                        float* __restrict__ out) {
  int bh = blockIdx.x;
  int b = bh / HH, h = bh % HH;
  __shared__ __align__(16) __bf16 aT[200 * 72];   // [padded_w][c], stride 72
  int tid = threadIdx.x;
  char* ldsb = (char*)&aT[0];

  // zero-fill pad rows 0..3 and 196..199 (cols 0..63), one b32 per thread
  {
    int r  = tid >> 5;                 // 0..7
    int cp = tid & 31;                 // column pair
    int row = (r < 4) ? r : (192 + r);
    *(uint32_t*)(ldsb + row * 144 + cp * 4) = 0u;
  }
  // async-stage 192 rows of a (BHWC bf16, contiguous) into padded rows 4..195
  const uint4* srcg = (const uint4*)(a_ws + (size_t)bh * (WW * CC));
  for (int q = tid; q < 1536; q += 256) {
    int wr = q >> 3, part = q & 7;
    uint32_t loff = (uint32_t)(uintptr_t)(ldsb + (wr + 4) * 144 + part * 16);
    const uint4* gp = srcg + q;
    asm volatile("global_load_async_to_lds_b128 %0, %1, off"
                 :: "v"(loff), "v"(gp) : "memory");
  }
  asm volatile("s_wait_asynccnt 0x0" ::: "memory");
  __syncthreads();

  int lane = tid & 31, wave = tid >> 5;
  int n = lane & 15, half = lane >> 4;
  int mt = wave >> 1;                  // wave owns one M-tile
  int ntbase = (wave & 1) * 6;         // six N-tiles

  // per-lane weight base; fragment (k,ks2) lives at byte offset
  // k*8192 + mt*2048 + ks2*1024 + lane*32
  const char* wbase = (const char*)wfrag + lane * 32 + mt * 2048;
  // LDS base: row (ntbase*16+n), channel-half offset; tile i at +i*2304
  const char* lb = ldsb + (ntbase * 16 + n) * 144 + half * 32;

  union BF16x16 { uint4 u[2]; v16bf v; };
  v8f a0 = {}, a1 = {}, a2 = {}, a3 = {}, a4 = {}, a5 = {};

  #pragma unroll 1                     // keep A loads inside the loop (no LICM blowup)
  for (int k = 0; k < TAPS; ++k) {
    const char* wk = wbase + k * 8192;
    const char* lk = lb + k * 144;
    #pragma unroll
    for (int ks2 = 0; ks2 < 2; ++ks2) {
      v16bf af = *(const v16bf*)(wk + ks2 * 1024);
      const char* sk = lk + ks2 * 64;
      BF16x16 b0, b1, b2f, b3, b4, b5;
      b0.u[0] = *(const uint4*)(sk);          b0.u[1] = *(const uint4*)(sk + 16);
      b1.u[0] = *(const uint4*)(sk + 2304);   b1.u[1] = *(const uint4*)(sk + 2320);
      b2f.u[0] = *(const uint4*)(sk + 4608);  b2f.u[1] = *(const uint4*)(sk + 4624);
      b3.u[0] = *(const uint4*)(sk + 6912);   b3.u[1] = *(const uint4*)(sk + 6928);
      b4.u[0] = *(const uint4*)(sk + 9216);   b4.u[1] = *(const uint4*)(sk + 9232);
      b5.u[0] = *(const uint4*)(sk + 11520);  b5.u[1] = *(const uint4*)(sk + 11536);
      a0 = __builtin_amdgcn_wmma_f32_16x16x32_bf16(false, af, false, b0.v,  (short)0, a0, false, false);
      a1 = __builtin_amdgcn_wmma_f32_16x16x32_bf16(false, af, false, b1.v,  (short)0, a1, false, false);
      a2 = __builtin_amdgcn_wmma_f32_16x16x32_bf16(false, af, false, b2f.v, (short)0, a2, false, false);
      a3 = __builtin_amdgcn_wmma_f32_16x16x32_bf16(false, af, false, b3.v,  (short)0, a3, false, false);
      a4 = __builtin_amdgcn_wmma_f32_16x16x32_bf16(false, af, false, b4.v,  (short)0, a4, false, false);
      a5 = __builtin_amdgcn_wmma_f32_16x16x32_bf16(false, af, false, b5.v,  (short)0, a5, false, false);
    }
  }

  // bias + residual + store  (C/D layout: VGPR r -> M = r + 8*half)
  #pragma unroll
  for (int r = 0; r < 8; ++r) {
    int o = mt * 16 + half * 8 + r;
    size_t row = (((size_t)b * CC + o) * HH + h) * WW + (size_t)(ntbase * 16 + n);
    float bias = b2[o];
    out[row]      = a0[r] + bias + x[row];
    out[row + 16] = a1[r] + bias + x[row + 16];
    out[row + 32] = a2[r] + bias + x[row + 32];
    out[row + 48] = a3[r] + bias + x[row + 48];
    out[row + 64] = a4[r] + bias + x[row + 64];
    out[row + 80] = a5[r] + bias + x[row + 80];
  }
}

// ---------------------------------------------------------------------------
extern "C" void kernel_launch(void* const* d_in, const int* in_sizes, int n_in,
                              void* d_out, int out_size, void* d_ws, size_t ws_size,
                              hipStream_t stream) {
  const float* x   = (const float*)d_in[0];
  const float* sw  = (const float*)d_in[1];
  const float* sb  = (const float*)d_in[2];
  const float* cw1 = (const float*)d_in[3];
  const float* cb1 = (const float*)d_in[4];
  const float* cw2 = (const float*)d_in[5];
  const float* cb2 = (const float*)d_in[6];
  const float* w2  = (const float*)d_in[7];
  const float* b2  = (const float*)d_in[8];
  float* out = (float*)d_out;

  char* ws = (char*)d_ws;
  float*  g     = (float*)(ws);                  // 256 f32
  float*  cf    = (float*)(ws + 1024);           // 2304 f32
  __bf16* wfrag = (__bf16*)(ws + 10240);         // 36864 bf16
  __bf16* a_ws  = (__bf16*)(ws + 83968);         // 9437184 bf16

  k_gap  <<<BB * CC, 256, 0, stream>>>(x, g);
  k_cf   <<<1,       256, 0, stream>>>(g, cw1, cb1, cw2, cb2, cf);
  k_wpack<<<144,     256, 0, stream>>>(w2, wfrag);
  k_ddf  <<<BB * HH, WW,  0, stream>>>(x, sw, sb, cf, a_ws);
  k_conv9<<<BB * HH, 256, 0, stream>>>(a_ws, wfrag, x, b2, out);
}
